// LSTM_35519379538276
// MI455X (gfx1250) — matrix operand order repaired and checked by reading, not compiled, not run on hardware
//
#include <hip/hip_runtime.h>

#define B_SZ   16384
#define IN_SZ  512
#define HID_SZ 512
#define KTOT   1024   // IN + HID

typedef __attribute__((ext_vector_type(16))) __bf16 v16bf;
typedef __attribute__((ext_vector_type(8)))  float  v8f;

union FragU { uint4 u[2]; v16bf v; };

__device__ __forceinline__ unsigned short f32_to_bf16(float f) {
  unsigned int u = __float_as_uint(f);
  u += 0x7FFFu + ((u >> 16) & 1u);   // round-to-nearest-even
  return (unsigned short)(u >> 16);
}

__device__ __forceinline__ unsigned int pack_bf16x2(float x, float y) {
#if __has_builtin(__builtin_amdgcn_cvt_pk_bf16_f32)
  typedef __attribute__((ext_vector_type(2))) __bf16 v2bf;
  v2bf r = __builtin_amdgcn_cvt_pk_bf16_f32(x, y);
  return __builtin_bit_cast(unsigned int, r);
#else
  return (unsigned int)f32_to_bf16(x) | ((unsigned int)f32_to_bf16(y) << 16);
#endif
}

__device__ __forceinline__ float fast_sigmoid(float x) {
  return 1.0f / (1.0f + __expf(-x));
}

__device__ __forceinline__ float fast_tanh(float x) {
  x = fminf(fmaxf(x, -12.0f), 12.0f);
  float e = __expf(2.0f * x);
  return (e - 1.0f) / (e + 1.0f);
}

// --- CDNA5 async LDS copy: 16B per lane, tracked by ASYNCcnt ----------------
__device__ __forceinline__ void async_copy_b128(unsigned lds_addr,
                                                const void* base,
                                                unsigned byte_off) {
  unsigned long long b64 = (unsigned long long)base;
  asm volatile("global_load_async_to_lds_b128 %0, %1, %2 offset:0"
               :: "v"(lds_addr), "v"(byte_off), "s"(b64)
               : "memory");
}

template <int N>
__device__ __forceinline__ void wait_asynccnt() {
  asm volatile("s_wait_asynccnt %0" :: "n"(N) : "memory");
}

__device__ __forceinline__ unsigned lds_off(const void* p) {
  return (unsigned)(size_t)p;   // addr[31:0] of a generic LDS pointer = LDS byte address
}

// A fragment: 16x32 bf16 tile. lane&15 = M row; lane>>4 selects K-half.
__device__ __forceinline__ v16bf frag_a(const unsigned short* row, int kc, int khalf) {
  FragU f;
  f.u[0] = *(const uint4*)(row + kc + khalf * 8);
  f.u[1] = *(const uint4*)(row + kc + 16 + khalf * 8);
  return f.v;
}

// B fragment: 32x16 bf16 tile. lane&15 = N column; lane>>4 selects K-half (same K order as A).
__device__ __forceinline__ v16bf frag_b(const unsigned short* row, int kc, int khalf) {
  FragU f;
  f.u[0] = *(const uint4*)(row + kc + khalf * 8);
  f.u[1] = *(const uint4*)(row + kc + 16 + khalf * 8);
  return f.v;
}

// ---------------------------------------------------------------------------
// Pass 0a: weights (gate order f,i,o,c) + V -> bf16.
// ---------------------------------------------------------------------------
__global__ void lstm_convert_weights(const float* __restrict__ Wf,
                                     const float* __restrict__ Wi,
                                     const float* __restrict__ Wo,
                                     const float* __restrict__ Wc,
                                     const float* __restrict__ Vw,
                                     unsigned short* __restrict__ dst) {
  const int WG  = HID_SZ * KTOT;                 // 524288
  const int TOT = 4 * WG + HID_SZ * HID_SZ;
  int idx = blockIdx.x * blockDim.x + threadIdx.x;
  if (idx >= TOT) return;
  float v;
  if      (idx <     WG) v = Wf[idx];
  else if (idx < 2 * WG) v = Wi[idx -     WG];
  else if (idx < 3 * WG) v = Wo[idx - 2 * WG];
  else if (idx < 4 * WG) v = Wc[idx - 3 * WG];
  else                   v = Vw[idx - 4 * WG];
  dst[idx] = f32_to_bf16(v);
}

// ---------------------------------------------------------------------------
// Pass 0b: combined = [inputs | hidden] -> bf16 [B][1024].
// ---------------------------------------------------------------------------
__global__ void lstm_convert_acts(const float* __restrict__ inputs,
                                  const float* __restrict__ hidden,
                                  unsigned short* __restrict__ dst) {
  int id = blockIdx.x * blockDim.x + threadIdx.x;   // one float4 chunk each
  int b  = id >> 8;                                  // 256 chunks per row
  int c  = id & 255;
  const float* src = (c < 128) ? (inputs + (size_t)b * IN_SZ + c * 4)
                               : (hidden + (size_t)b * HID_SZ + (c - 128) * 4);
  float4 s = *(const float4*)src;
  *(uint2*)(dst + (size_t)b * KTOT + c * 4) =
      make_uint2(pack_bf16x2(s.x, s.y), pack_bf16x2(s.z, s.w));
}

// ---------------------------------------------------------------------------
// Pass 1: fused 4-gate GEMM + LSTM pointwise update.
// Grid: (B/128, HID/32). Block: 256 threads = 8 waves.
// Double-buffered LDS staged with global_load_async_to_lds_b128.
// ---------------------------------------------------------------------------
__launch_bounds__(256)
__global__ void lstm_gates_kernel(const unsigned short* __restrict__ Abf, // [B][1024] bf16
                                  const float* __restrict__ cell,
                                  const unsigned short* __restrict__ Wbf, // [4][512][1024] bf16
                                  const float* __restrict__ bF,
                                  const float* __restrict__ bI,
                                  const float* __restrict__ bO,
                                  const float* __restrict__ bC,
                                  float* __restrict__ out_cell,
                                  float* __restrict__ out_hidden,
                                  unsigned short* __restrict__ nh_bf) {   // [B][512] bf16
  __shared__ __align__(16) unsigned short As[2][128][72];
  __shared__ __align__(16) unsigned short Bs[2][4][32][72];

  const int tid  = threadIdx.x;
  const int lane = tid & 31;
  const int wid  = tid >> 5;
  const int mb   = wid & 3;
  const int hb   = wid >> 2;
  const int ln   = lane & 15;
  const int lh   = lane >> 4;

  const int b0 = blockIdx.x * 128;
  const int h0 = blockIdx.y * 32;

  // Per-thread staging coordinates (4 A chunks + 4 B chunks per stage).
  const int ar = tid >> 1;                 // unused placeholder removed below
  (void)ar;

  v8f acc[2][4] = {};

  const int NS = KTOT / 64;                // 16 stages

  // --- stage issue: 8 async b128 copies per thread --------------------------
  auto issue = [&](int ks, int buf) {
    const int k0 = ks * 64;
    #pragma unroll
    for (int i = 0; i < 4; ++i) {          // A: 1024 chunks / 256 thr
      int id = tid + i * 256;
      int r  = id >> 3;                    // 0..127
      int c  = id & 7;
      unsigned goff = ((unsigned)(b0 + r) * KTOT + (unsigned)(k0 + c * 8)) * 2u;
      async_copy_b128(lds_off(&As[buf][r][c * 8]), Abf, goff);
    }
    #pragma unroll
    for (int i = 0; i < 4; ++i) {          // B: 4 gates x 32 x 8 chunks
      int id = tid + i * 256;
      int g  = id >> 8;
      int r  = (id >> 3) & 31;
      int c  = id & 7;
      unsigned goff = (((unsigned)g * HID_SZ + (unsigned)(h0 + r)) * KTOT +
                       (unsigned)(k0 + c * 8)) * 2u;
      async_copy_b128(lds_off(&Bs[buf][g][r][c * 8]), Wbf, goff);
    }
  };

  issue(0, 0);

  for (int ks = 0; ks < NS; ++ks) {
    const int cur = ks & 1;
    if (ks + 1 < NS) {
      issue(ks + 1, cur ^ 1);
      wait_asynccnt<8>();                  // stage ks complete; ks+1 in flight
    } else {
      wait_asynccnt<0>();
    }
    __syncthreads();

    const unsigned short* arow0 = &As[cur][mb * 32 + ln][0];
    const unsigned short* arow1 = arow0 + 16 * 72;
    const unsigned short* brow  = &Bs[cur][0][hb * 16 + ln][0];
    #pragma unroll
    for (int kc = 0; kc < 64; kc += 32) {
      v16bf a0 = frag_a(arow0, kc, lh);
      v16bf a1 = frag_a(arow1, kc, lh);
      #pragma unroll
      for (int g = 0; g < 4; ++g) {
        v16bf bb = frag_b(brow + g * 32 * 72, kc, lh);
        acc[0][g] = __builtin_amdgcn_wmma_f32_16x16x32_bf16(
            false, a0, false, bb, (short)0, acc[0][g], false, false);
        acc[1][g] = __builtin_amdgcn_wmma_f32_16x16x32_bf16(
            false, a1, false, bb, (short)0, acc[1][g], false, false);
      }
    }
    __syncthreads();                       // all waves done reading buf[cur]
  }

  // Epilogue: C layout -> vgpr v, lanes 0-15 M=v, lanes 16-31 M=v+8; N=lane&15.
  const int h = h0 + hb * 16 + ln;
  const float bfv = bF[h], biv = bI[h], bov = bO[h], bcv = bC[h];
  #pragma unroll
  for (int t = 0; t < 2; ++t) {
    #pragma unroll
    for (int v = 0; v < 8; ++v) {
      int b = b0 + mb * 32 + t * 16 + lh * 8 + v;
      float gf = fast_sigmoid(acc[t][0][v] + bfv);
      float gi = fast_sigmoid(acc[t][1][v] + biv);
      float go = fast_sigmoid(acc[t][2][v] + bov);
      float gc = fast_tanh  (acc[t][3][v] + bcv);
      float cs = cell[(size_t)b * HID_SZ + h];
      float nc = gf * cs + gi * gc;
      float nh = go * fast_tanh(nc);
      out_cell  [(size_t)b * HID_SZ + h] = nc;
      out_hidden[(size_t)b * HID_SZ + h] = nh;
      nh_bf     [(size_t)b * HID_SZ + h] = f32_to_bf16(nh);
    }
  }
}

// ---------------------------------------------------------------------------
// Pass 2: output = new_hidden @ V^T + V_b.
// Grid: (B/128, OUT/64). Block: 256 threads = 8 waves. Async double-buffered.
// ---------------------------------------------------------------------------
__launch_bounds__(256)
__global__ void lstm_out_kernel(const unsigned short* __restrict__ nh_bf, // [B][512]
                                const unsigned short* __restrict__ Vbf,   // [512][512]
                                const float* __restrict__ Vb,
                                float* __restrict__ out) {
  __shared__ __align__(16) unsigned short As[2][128][72];
  __shared__ __align__(16) unsigned short Bs[2][64][72];

  const int tid  = threadIdx.x;
  const int lane = tid & 31;
  const int wid  = tid >> 5;
  const int mb   = wid & 3;
  const int nb   = wid >> 2;
  const int ln   = lane & 15;
  const int lh   = lane >> 4;

  const int b0 = blockIdx.x * 128;
  const int n0 = blockIdx.y * 64;

  v8f acc[2][2] = {};
  const int NS = HID_SZ / 64;              // 8 stages

  auto issue = [&](int ks, int buf) {
    const int k0 = ks * 64;
    #pragma unroll
    for (int i = 0; i < 4; ++i) {          // A: 1024 chunks
      int id = tid + i * 256;
      int r  = id >> 3;
      int c  = id & 7;
      unsigned goff = ((unsigned)(b0 + r) * HID_SZ + (unsigned)(k0 + c * 8)) * 2u;
      async_copy_b128(lds_off(&As[buf][r][c * 8]), nh_bf, goff);
    }
    #pragma unroll
    for (int i = 0; i < 2; ++i) {          // B: 512 chunks
      int id = tid + i * 256;
      int r  = id >> 3;                    // 0..63
      int c  = id & 7;
      unsigned goff = ((unsigned)(n0 + r) * HID_SZ + (unsigned)(k0 + c * 8)) * 2u;
      async_copy_b128(lds_off(&Bs[buf][r][c * 8]), Vbf, goff);
    }
  };

  issue(0, 0);

  for (int ks = 0; ks < NS; ++ks) {
    const int cur = ks & 1;
    if (ks + 1 < NS) {
      issue(ks + 1, cur ^ 1);
      wait_asynccnt<6>();
    } else {
      wait_asynccnt<0>();
    }
    __syncthreads();

    const unsigned short* arow0 = &As[cur][mb * 32 + ln][0];
    const unsigned short* arow1 = arow0 + 16 * 72;
    #pragma unroll
    for (int kc = 0; kc < 64; kc += 32) {
      v16bf a0 = frag_a(arow0, kc, lh);
      v16bf a1 = frag_a(arow1, kc, lh);
      #pragma unroll
      for (int u = 0; u < 2; ++u) {
        v16bf bb = frag_b(&Bs[cur][nb * 32 + u * 16 + ln][0], kc, lh);
        acc[0][u] = __builtin_amdgcn_wmma_f32_16x16x32_bf16(
            false, a0, false, bb, (short)0, acc[0][u], false, false);
        acc[1][u] = __builtin_amdgcn_wmma_f32_16x16x32_bf16(
            false, a1, false, bb, (short)0, acc[1][u], false, false);
      }
    }
    __syncthreads();
  }

  #pragma unroll
  for (int u = 0; u < 2; ++u) {
    const int n = n0 + nb * 32 + u * 16 + ln;
    const float bias = Vb[n];
    #pragma unroll
    for (int t = 0; t < 2; ++t) {
      #pragma unroll
      for (int v = 0; v < 8; ++v) {
        int b = b0 + mb * 32 + t * 16 + lh * 8 + v;
        out[(size_t)b * HID_SZ + n] = acc[t][u][v] + bias;
      }
    }
  }
}

// ---------------------------------------------------------------------------
extern "C" void kernel_launch(void* const* d_in, const int* in_sizes, int n_in,
                              void* d_out, int out_size, void* d_ws, size_t ws_size,
                              hipStream_t stream) {
  const float* inputs = (const float*)d_in[0];
  const float* cell   = (const float*)d_in[1];
  const float* hidden = (const float*)d_in[2];
  const float* Wf_w   = (const float*)d_in[3];
  const float* Wf_b   = (const float*)d_in[4];
  const float* Wi_w   = (const float*)d_in[5];
  const float* Wi_b   = (const float*)d_in[6];
  const float* Wc_w   = (const float*)d_in[7];
  const float* Wc_b   = (const float*)d_in[8];
  const float* Wo_w   = (const float*)d_in[9];
  const float* Wo_b   = (const float*)d_in[10];
  const float* V_w    = (const float*)d_in[11];
  const float* V_b    = (const float*)d_in[12];

  // Workspace layout (bf16):
  unsigned short* Wbf   = (unsigned short*)d_ws;                    // 4*512*1024
  unsigned short* Vbf   = Wbf + (size_t)4 * HID_SZ * KTOT;          // 512*512
  unsigned short* Abf   = Vbf + (size_t)HID_SZ * HID_SZ;            // B*1024
  unsigned short* nh_bf = Abf + (size_t)B_SZ * KTOT;                // B*512

  float* out_cell   = (float*)d_out;
  float* out_hidden = out_cell   + (size_t)B_SZ * HID_SZ;
  float* out_y      = out_hidden + (size_t)B_SZ * HID_SZ;

  const int totW = 4 * HID_SZ * KTOT + HID_SZ * HID_SZ;
  lstm_convert_weights<<<(totW + 255) / 256, 256, 0, stream>>>(
      Wf_w, Wi_w, Wo_w, Wc_w, V_w, Wbf);

  lstm_convert_acts<<<(B_SZ * 256) / 256, 256, 0, stream>>>(inputs, hidden, Abf);

  lstm_gates_kernel<<<dim3(B_SZ / 128, HID_SZ / 32), 256, 0, stream>>>(
      Abf, cell, Wbf, Wf_b, Wi_b, Wo_b, Wc_b, out_cell, out_hidden, nh_bf);

  lstm_out_kernel<<<dim3(B_SZ / 128, HID_SZ / 64), 256, 0, stream>>>(
      nh_bf, Vbf, V_b, out_y);
}
